// BERT_12661563588899
// MI455X (gfx1250) — matrix operand order repaired and checked by reading, not compiled, not run on hardware
//
#include <hip/hip_runtime.h>

// ---------------------------------------------------------------------------
// BERT forward for MI455X (gfx1250): bf16 WMMA GEMMs, f32 accumulation,
// async global->LDS staging (ASYNCcnt), all weights pre-transposed to [N,K].
// ---------------------------------------------------------------------------

typedef __attribute__((ext_vector_type(16))) __bf16 v16bf;
typedef __attribute__((ext_vector_type(8)))  __bf16 v8bf;
typedef __attribute__((ext_vector_type(8)))  float  v8f;
typedef __attribute__((ext_vector_type(4)))  int    v4i;

#define ASG __attribute__((address_space(1)))
#define ASL __attribute__((address_space(3)))

#define BB   4
#define SS   1024
#define DD   768
#define HH   4
#define DQQ  192
#define DFFQ 3072
#define VOC  30522

// ------------------------------- converts ----------------------------------

// generic fp32 [K,N] -> bf16 transposed [N,K]
__global__ __launch_bounds__(256) void k_convT(const float* __restrict__ s,
                                               __bf16* __restrict__ d,
                                               int K, int N) {
  long long i = (long long)blockIdx.x * 256 + threadIdx.x;
  long long n = (long long)K * N;
  if (i >= n) return;
  int k = (int)(i / N), c = (int)(i % N);
  d[(size_t)c * K + k] = (__bf16)s[i];
}

// [H,D,DQ] fp32 -> bf16 transposed [H*DQ, D]
__global__ __launch_bounds__(256) void k_convert_qkvT(const float* __restrict__ s,
                                                      __bf16* __restrict__ d) {
  int i = blockIdx.x * 256 + threadIdx.x;
  int n = HH * DD * DQQ;
  if (i >= n) return;
  int h = i / (DD * DQQ);
  int r = i % (DD * DQQ);
  int dd = r / DQQ, j = r % DQQ;
  d[(size_t)(h * DQQ + j) * DD + dd] = (__bf16)s[i];
}

// --------------------------- embed + layernorm -----------------------------

__global__ __launch_bounds__(256) void k_embed_ln(const int* __restrict__ ids,
                                                  const float* __restrict__ tok,
                                                  const float* __restrict__ seg,
                                                  const float* __restrict__ pos,
                                                  const float* __restrict__ g,
                                                  const float* __restrict__ bt,
                                                  __bf16* __restrict__ xout) {
  int row = blockIdx.x;            // b*S + s
  int s   = row & (SS - 1);
  int id  = ids[row];
  int tid = threadIdx.x, lane = tid & 31, wv = tid >> 5;
  float vals[3], sum = 0.f, sq = 0.f;
#pragma unroll
  for (int j = 0; j < 3; ++j) {
    int d = tid + j * 256;
    float v = tok[(size_t)id * DD + d] + seg[(size_t)id * DD + d] +
              pos[(size_t)s * DD + d];
    vals[j] = v; sum += v; sq += v * v;
  }
#pragma unroll
  for (int o = 16; o > 0; o >>= 1) {
    sum += __shfl_xor(sum, o, 32);
    sq  += __shfl_xor(sq,  o, 32);
  }
  __shared__ float s1[8], s2[8];
  if (lane == 0) { s1[wv] = sum; s2[wv] = sq; }
  __syncthreads();
  float ts = 0.f, tq = 0.f;
#pragma unroll
  for (int i = 0; i < 8; ++i) { ts += s1[i]; tq += s2[i]; }
  float mu  = ts / DD;
  float var = tq / DD - mu * mu;
  float inv = rsqrtf(var + 1e-5f);
#pragma unroll
  for (int j = 0; j < 3; ++j) {
    int d = tid + j * 256;
    xout[(size_t)row * DD + d] = (__bf16)((vals[j] - mu) * inv * g[d] + bt[d]);
  }
}

// ------------------------------ WMMA GEMM ----------------------------------

struct GemmP {
  const __bf16* A; const __bf16* B; void* C; const float* bias;
  long long sAo, sAi, sBo, sBi, sCo, sCi;   // z = zo*inner + zi strides (elems)
  int M, N, K, lda, ldb, ldc, zInner, flags;
  float alpha;
};
enum { GF_BT = 1, GF_RELU = 2, GF_OUTBF = 4 };

static __device__ inline v8bf v8bf_zero() {
  v8bf z;
#pragma unroll
  for (int i = 0; i < 8; ++i) z[i] = (__bf16)0.0f;
  return z;
}

__global__ __launch_bounds__(256) void k_gemm(GemmP p) {
  __shared__ __attribute__((aligned(16))) __bf16 At[128][40];
  __shared__ __attribute__((aligned(16))) __bf16 Bt[128][40];

  int z  = blockIdx.z;
  int zo = z / p.zInner, zi = z % p.zInner;
  const __bf16* A  = p.A + zo * p.sAo + zi * p.sAi;
  const __bf16* Bp = p.B + zo * p.sBo + zi * p.sBi;

  int m0 = blockIdx.y * 128, n0 = blockIdx.x * 128;
  int tid = threadIdx.x, lane = tid & 31, wv = tid >> 5;
  int wm = wv >> 2, wn = wv & 3;              // 2x4 wave grid: 64 rows x 32 cols
  int rlo = lane & 15;

  const bool fullA = (m0 + 128 <= p.M);
  const bool fullB = (n0 + 128 <= p.N);

  v8f acc[4][2] = {};

  for (int k0 = 0; k0 < p.K; k0 += 32) {
    // ---- stage A tile [128 x 32] ----
    if (fullA) {
      // async DMA: global -> LDS, tracked by ASYNCcnt, no VGPR round-trip
#pragma unroll
      for (int it = 0; it < 2; ++it) {
        int idx = tid + it * 256;
        int r = idx >> 2, c = idx & 3;
        __builtin_amdgcn_global_load_async_to_lds_b128(
            (ASG v4i*)(A + (size_t)(m0 + r) * p.lda + k0 + c * 8),
            (ASL v4i*)&At[r][c * 8], 0, 0);
      }
    } else {
#pragma unroll
      for (int it = 0; it < 2; ++it) {
        int idx = tid + it * 256;
        int r = idx >> 2, c = idx & 3;
        v8bf val = v8bf_zero();
        int gm = m0 + r;
        if (gm < p.M)
          val = *(const v8bf*)(A + (size_t)gm * p.lda + k0 + c * 8);
        *(v8bf*)&At[r][c * 8] = val;
      }
    }
    // ---- stage B tile transposed in LDS: Bt[n][k] ----
    if (p.flags & GF_BT) {        // B given as [N,K] row-major
      if (fullB) {
#pragma unroll
        for (int it = 0; it < 2; ++it) {
          int idx = tid + it * 256;
          int r = idx >> 2, c = idx & 3;
          __builtin_amdgcn_global_load_async_to_lds_b128(
              (ASG v4i*)(Bp + (size_t)(n0 + r) * p.ldb + k0 + c * 8),
              (ASL v4i*)&Bt[r][c * 8], 0, 0);
        }
      } else {
        if (k0 + 32 < p.K)      // keep next tile warm for the edge blocks
          __builtin_prefetch(Bp + (size_t)(n0 + (tid >> 2)) * p.ldb + k0 + 32,
                             0, 1);
#pragma unroll
        for (int it = 0; it < 2; ++it) {
          int idx = tid + it * 256;
          int r = idx >> 2, c = idx & 3;
          v8bf val = v8bf_zero();
          int gn = n0 + r;
          if (gn < p.N)
            val = *(const v8bf*)(Bp + (size_t)gn * p.ldb + k0 + c * 8);
          *(v8bf*)&Bt[r][c * 8] = val;
        }
      }
    } else {                      // B given as [K,N] row-major -> transpose
#pragma unroll
      for (int it = 0; it < 2; ++it) {
        int idx = tid + it * 256;
        int k = idx >> 4, nc = idx & 15;
        int gk = k0 + k;
#pragma unroll
        for (int e = 0; e < 8; ++e) {
          int gn = n0 + nc * 8 + e;
          __bf16 v = (__bf16)0.0f;
          if (gk < p.K && gn < p.N) v = Bp[(size_t)gk * p.ldb + gn];
          Bt[nc * 8 + e][k] = v;
        }
      }
    }
    // drain our async copies, then make all waves' LDS writes visible
    asm volatile("s_wait_asynccnt 0x0" ::: "memory");
    __syncthreads();

    // ---- fragment loads (per ISA 16-bit A 16x32 / B 32x16 layouts) ----
    v16bf af[4], bfr[2];
    int akb = (lane < 16) ? 0 : 8;
#pragma unroll
    for (int mi = 0; mi < 4; ++mi) {
      int r = wm * 64 + mi * 16 + rlo;
      v8bf lo = *(const v8bf*)&At[r][akb];
      v8bf hi = *(const v8bf*)&At[r][akb + 16];
#pragma unroll
      for (int e = 0; e < 8; ++e) { af[mi][e] = lo[e]; af[mi][e + 8] = hi[e]; }
    }
    int bkb = (lane < 16) ? 0 : 8;
#pragma unroll
    for (int ni = 0; ni < 2; ++ni) {
      int cc = wn * 32 + ni * 16 + rlo;
      v8bf lo = *(const v8bf*)&Bt[cc][bkb];
      v8bf hi = *(const v8bf*)&Bt[cc][bkb + 16];
#pragma unroll
      for (int e = 0; e < 8; ++e) { bfr[ni][e] = lo[e]; bfr[ni][e + 8] = hi[e]; }
    }

#pragma unroll
    for (int mi = 0; mi < 4; ++mi)
#pragma unroll
      for (int ni = 0; ni < 2; ++ni)
        acc[mi][ni] = __builtin_amdgcn_wmma_f32_16x16x32_bf16(
            false, af[mi], false, bfr[ni], (short)0, acc[mi][ni], false, false);

    __syncthreads();
  }

  // ---- epilogue: alpha, bias, relu, store f32 or bf16 ----
  bool outbf = p.flags & GF_OUTBF;
  bool relu  = p.flags & GF_RELU;
  __bf16* Cb = (__bf16*)p.C + zo * p.sCo + zi * p.sCi;
  float*  Cf = (float*)p.C + zo * p.sCo + zi * p.sCi;
#pragma unroll
  for (int mi = 0; mi < 4; ++mi) {
#pragma unroll
    for (int ni = 0; ni < 2; ++ni) {
      int n = n0 + wn * 32 + ni * 16 + rlo;
      if (n >= p.N) continue;
      float bs = p.bias ? p.bias[n] : 0.0f;
      int mb = m0 + wm * 64 + mi * 16 + ((lane >> 4) << 3);
      v8f c = acc[mi][ni];
#pragma unroll
      for (int i = 0; i < 8; ++i) {
        int m = mb + i;
        if (m < p.M) {
          float v = c[i] * p.alpha + bs;
          if (relu) v = v > 0.f ? v : 0.f;
          if (outbf) Cb[(size_t)m * p.ldc + n] = (__bf16)v;
          else       Cf[(size_t)m * p.ldc + n] = v;
        }
      }
    }
  }
}

// ------------------------------- softmax -----------------------------------

__global__ __launch_bounds__(256) void k_softmax(const float* __restrict__ scores,
                                                 const unsigned char* __restrict__ mask,
                                                 __bf16* __restrict__ probs) {
  int row = blockIdx.x;            // (b*H + h)*S + s
  int s  = row & (SS - 1);
  int bh = row >> 10;
  int b  = bh >> 2;
  const float* sr = scores + (size_t)row * SS;
  const unsigned char* mr = mask + ((size_t)b * SS + s) * SS;
  int tid = threadIdx.x, lane = tid & 31, wv = tid >> 5;
  __shared__ float sh[8];

  float v[4], mx = -1e30f;
#pragma unroll
  for (int j = 0; j < 4; ++j) {
    int t = tid + j * 256;
    float x = sr[t];
    if (mr[t]) x = -1e9f;
    v[j] = x; mx = fmaxf(mx, x);
  }
#pragma unroll
  for (int o = 16; o > 0; o >>= 1) mx = fmaxf(mx, __shfl_xor(mx, o, 32));
  if (lane == 0) sh[wv] = mx;
  __syncthreads();
  float rmx = sh[0];
#pragma unroll
  for (int i = 1; i < 8; ++i) rmx = fmaxf(rmx, sh[i]);
  __syncthreads();

  float sum = 0.f;
#pragma unroll
  for (int j = 0; j < 4; ++j) { v[j] = __expf(v[j] - rmx); sum += v[j]; }
#pragma unroll
  for (int o = 16; o > 0; o >>= 1) sum += __shfl_xor(sum, o, 32);
  if (lane == 0) sh[wv] = sum;
  __syncthreads();
  float tot = 0.f;
#pragma unroll
  for (int i = 0; i < 8; ++i) tot += sh[i];
  float inv = 1.0f / tot;
#pragma unroll
  for (int j = 0; j < 4; ++j) {
    int t = tid + j * 256;
    probs[(size_t)row * SS + t] = (__bf16)(v[j] * inv);
  }
}

// --------------------------------- cls -------------------------------------

__global__ __launch_bounds__(256) void k_cls(const __bf16* __restrict__ ff,
                                             const float* __restrict__ Wcls,
                                             const float* __restrict__ bcls,
                                             float* __restrict__ out) {
  int wv = threadIdx.x >> 5, lane = threadIdx.x & 31;
  int b = wv >> 1, c = wv & 1;
  const __bf16* row = ff + (size_t)b * SS * DD;   // token 0 of batch b
  float sum = 0.f;
  for (int d = lane; d < DD; d += 32)
    sum += (float)row[d] * Wcls[(size_t)d * 2 + c];
#pragma unroll
  for (int o = 16; o > 0; o >>= 1) sum += __shfl_xor(sum, o, 32);
  if (lane == 0) out[wv] = sum + bcls[c];
}

// ------------------------------- host side ---------------------------------

static void launch_gemm(const __bf16* A, const __bf16* B, void* C, const float* bias,
                        int M, int N, int K, int lda, int ldb, int ldc,
                        int flags, float alpha, int batches, int zInner,
                        long long sAo, long long sAi, long long sBo, long long sBi,
                        long long sCo, long long sCi, hipStream_t st) {
  GemmP p;
  p.A = A; p.B = B; p.C = C; p.bias = bias;
  p.sAo = sAo; p.sAi = sAi; p.sBo = sBo; p.sBi = sBi; p.sCo = sCo; p.sCi = sCi;
  p.M = M; p.N = N; p.K = K; p.lda = lda; p.ldb = ldb; p.ldc = ldc;
  p.zInner = zInner; p.flags = flags; p.alpha = alpha;
  dim3 grid((N + 127) / 128, (M + 127) / 128, batches);
  k_gemm<<<grid, 256, 0, st>>>(p);
}

extern "C" void kernel_launch(void* const* d_in, const int* in_sizes, int n_in,
                              void* d_out, int out_size, void* d_ws, size_t ws_size,
                              hipStream_t stream) {
  (void)in_sizes; (void)n_in; (void)out_size; (void)ws_size;
  const int*           ids  = (const int*)d_in[0];
  const unsigned char* mask = (const unsigned char*)d_in[1];
  const float* tok_emb = (const float*)d_in[2];
  const float* seg_emb = (const float*)d_in[3];
  const float* pos_emb = (const float*)d_in[4];
  const float* ln_g = (const float*)d_in[5];
  const float* ln_b = (const float*)d_in[6];
  const float* Wq = (const float*)d_in[7];  const float* bq = (const float*)d_in[8];
  const float* Wk = (const float*)d_in[9];  const float* bk = (const float*)d_in[10];
  const float* Wv = (const float*)d_in[11]; const float* bv = (const float*)d_in[12];
  const float* Wo = (const float*)d_in[13]; const float* bo = (const float*)d_in[14];
  const float* W1 = (const float*)d_in[15]; const float* b1 = (const float*)d_in[16];
  const float* W2 = (const float*)d_in[17]; const float* b2 = (const float*)d_in[18];
  const float* Wtok = (const float*)d_in[19]; const float* btok = (const float*)d_in[20];
  const float* Wcls = (const float*)d_in[21]; const float* bcls = (const float*)d_in[22];
  float* out = (float*)d_out;

  const int T = BB * SS;                 // 4096 tokens
  char* ws = (char*)d_ws;
  size_t off = 0;
  auto alloc = [&](size_t nbytes) -> char* {
    char* p = ws + off;
    off += (nbytes + 255) & ~(size_t)255;
    return p;
  };
  __bf16* x_bf    = (__bf16*)alloc((size_t)T * DD * 2);
  __bf16* WqT     = (__bf16*)alloc((size_t)DD * DD * 2);     // [HDQ, D]
  __bf16* WkT     = (__bf16*)alloc((size_t)DD * DD * 2);
  __bf16* WvT     = (__bf16*)alloc((size_t)DD * DD * 2);
  __bf16* WoT     = (__bf16*)alloc((size_t)DD * DD * 2);     // [D, HDQ]^T
  __bf16* W1T     = (__bf16*)alloc((size_t)DFFQ * DD * 2);   // [DFF, D]
  __bf16* W2T     = (__bf16*)alloc((size_t)DD * DFFQ * 2);   // [D, DFF]
  __bf16* WtokT   = (__bf16*)alloc((size_t)VOC * DD * 2);    // [VOC, D]
  __bf16* q_bf    = (__bf16*)alloc((size_t)T * DD * 2);
  __bf16* k_bf    = (__bf16*)alloc((size_t)T * DD * 2);
  __bf16* v_bf    = (__bf16*)alloc((size_t)T * DD * 2);
  float*  scores  = (float*)alloc((size_t)BB * HH * SS * SS * 4);
  __bf16* probs   = (__bf16*)alloc((size_t)BB * HH * SS * SS * 2);
  __bf16* ctx_bf  = (__bf16*)alloc((size_t)T * DD * 2);
  __bf16* ao_bf   = (__bf16*)alloc((size_t)T * DD * 2);
  __bf16* ff1_bf  = (__bf16*)alloc((size_t)T * DFFQ * 2);
  __bf16* ff_bf   = (__bf16*)alloc((size_t)T * DD * 2);

  // one-time weight conversion: fp32 -> bf16, transposed to [N,K]
  int nqkv = HH * DD * DQQ;
  k_convert_qkvT<<<(nqkv + 255) / 256, 256, 0, stream>>>(Wq, WqT);
  k_convert_qkvT<<<(nqkv + 255) / 256, 256, 0, stream>>>(Wk, WkT);
  k_convert_qkvT<<<(nqkv + 255) / 256, 256, 0, stream>>>(Wv, WvT);
  k_convT<<<(DD * DD + 255) / 256, 256, 0, stream>>>(Wo, WoT, DD, DD);
  k_convT<<<(DD * DFFQ + 255) / 256, 256, 0, stream>>>(W1, W1T, DD, DFFQ);
  k_convT<<<(DFFQ * DD + 255) / 256, 256, 0, stream>>>(W2, W2T, DFFQ, DD);
  long long ntok = (long long)DD * VOC;
  k_convT<<<(unsigned)((ntok + 255) / 256), 256, 0, stream>>>(Wtok, WtokT, DD, VOC);

  // embeddings + layernorm -> bf16 activations
  k_embed_ln<<<T, 256, 0, stream>>>(ids, tok_emb, seg_emb, pos_emb, ln_g, ln_b, x_bf);

  // QKV projections: [4096,768] x [768,768]^T (+bias)
  launch_gemm(x_bf, WqT, q_bf, bq, T, DD, DD, DD, DD, DD, GF_BT | GF_OUTBF, 1.0f,
              1, 1, 0, 0, 0, 0, 0, 0, stream);
  launch_gemm(x_bf, WkT, k_bf, bk, T, DD, DD, DD, DD, DD, GF_BT | GF_OUTBF, 1.0f,
              1, 1, 0, 0, 0, 0, 0, 0, stream);
  launch_gemm(x_bf, WvT, v_bf, bv, T, DD, DD, DD, DD, DD, GF_BT | GF_OUTBF, 1.0f,
              1, 1, 0, 0, 0, 0, 0, 0, stream);

  // scores = Q K^T / sqrt(S): z = b*H + h (16 batches), B transposed
  launch_gemm(q_bf, k_bf, scores, nullptr, SS, SS, DQQ, DD, DD, SS,
              GF_BT, 1.0f / 32.0f, BB * HH, HH,
              (long long)SS * DD, DQQ,                 // A: b, h strides
              (long long)SS * DD, DQQ,                 // B: b, h strides
              (long long)HH * SS * SS, (long long)SS * SS, stream);

  // softmax rows (mask applied) -> bf16 probs
  k_softmax<<<BB * HH * SS, 256, 0, stream>>>(scores, mask, probs);

  // ctx = P V: z = b*H + h, result packed as [t, h*DQ + j]  (B is [K,N])
  launch_gemm(probs, v_bf, ctx_bf, nullptr, SS, DQQ, SS, SS, DD, DD,
              GF_OUTBF, 1.0f, BB * HH, HH,
              (long long)HH * SS * SS, (long long)SS * SS,
              (long long)SS * DD, DQQ,
              (long long)SS * DD, DQQ, stream);

  // attn_out = ctx @ Wo + bo
  launch_gemm(ctx_bf, WoT, ao_bf, bo, T, DD, DD, DD, DD, DD, GF_BT | GF_OUTBF,
              1.0f, 1, 1, 0, 0, 0, 0, 0, 0, stream);

  // ff1 = relu(attn_out @ W1 + b1)
  launch_gemm(ao_bf, W1T, ff1_bf, b1, T, DFFQ, DD, DD, DD, DFFQ,
              GF_BT | GF_OUTBF | GF_RELU, 1.0f, 1, 1, 0, 0, 0, 0, 0, 0, stream);

  // ff = ff1 @ W2 + b2
  launch_gemm(ff1_bf, W2T, ff_bf, b2, T, DD, DFFQ, DFFQ, DFFQ, DD,
              GF_BT | GF_OUTBF, 1.0f, 1, 1, 0, 0, 0, 0, 0, 0, stream);

  // cls_out = ff[:,0,:] @ Wcls + bcls  -> out[0..7]
  k_cls<<<1, 256, 0, stream>>>(ff_bf, Wcls, bcls, out);

  // tok_out = ff @ Wtok + btok -> out[8..] (fp32, the 192-GFLOP GEMM)
  launch_gemm(ff_bf, WtokT, out + 8, btok, T, VOC, DD, DD, DD, VOC,
              GF_BT, 1.0f, 1, 1, 0, 0, 0, 0, 0, 0, stream);
}
